// RuleMiner_55250459296137
// MI455X (gfx1250) — matrix operand order repaired and checked by reading, not compiled, not run on hardware
//
#include <hip/hip_runtime.h>
#include <hip/hip_bf16.h>

#define RANK_      3
#define NSTEPS     3
#define NENT       40000
#define NOPS       12
#define DIM_       128
#define HID_       128
#define BATCH_     128
#define NNZ_       300000
#define G4H        (4 * HID_)          // 512
#define NOP1       (NOPS + 1)          // 13

typedef __attribute__((ext_vector_type(16))) _Float16 v16h;
typedef __attribute__((ext_vector_type(8)))  float    v8f;

#if __has_builtin(__builtin_amdgcn_global_load_async_to_lds_b32) && \
    __has_builtin(__builtin_amdgcn_s_wait_asynccnt)
#define HAVE_ASYNC_LDS 1
typedef __attribute__((address_space(1))) int* gptr_i32;
typedef __attribute__((address_space(3))) int* lptr_i32;
#else
#define HAVE_ASYNC_LDS 0
#endif

// ---------------------------------------------------------------------------
// WMMA fragment loaders (16x16x32 f16; this code's contiguous k order, same in A and B)
// ---------------------------------------------------------------------------
// A (16x32, MxK): lane<16 -> M=lane, K=0..15; lane>=16 -> K=16..31
__device__ __forceinline__ v16h load_a_frag_f32(const float* A, int lda, int m0,
                                                int k0, int lane) {
  v16h a;
  const int half = lane >> 4;
  const int m    = m0 + (lane & 15);
  #pragma unroll
  for (int e = 0; e < 16; ++e) {
    const int k = k0 + (half << 4) + e;
    a[e] = (_Float16)A[(size_t)m * lda + k];
  }
  return a;
}

// B (32x16, KxN) built from row-major W (N rows, K cols): B[k][n] = W[n][k]
// lanes 0-15 hold K=0..15 (N=lane), lanes 16-31 hold K=16..31 (N=lane-16)
__device__ __forceinline__ v16h load_bT_frag_f32(const float* W, int ldw, int n0,
                                                 int k0, int lane) {
  v16h b;
  const int n  = n0 + (lane & 15);
  const int kb = k0 + ((lane >> 4) << 4);
  #pragma unroll
  for (int e = 0; e < 16; ++e)
    b[e] = (_Float16)W[(size_t)n * ldw + kb + e];
  return b;
}

__device__ __forceinline__ float sigmoidf_(float x) { return 1.0f / (1.0f + __expf(-x)); }

// ---------------------------------------------------------------------------
// Small utility kernels
// ---------------------------------------------------------------------------
__global__ void zero4_kernel(float4* __restrict__ p, int n4) {
  int i = blockIdx.x * blockDim.x + threadIdx.x;
  if (i < n4) p[i] = make_float4(0.f, 0.f, 0.f, 0.f);
}

__global__ void gather_x_kernel(const float* __restrict__ emb,
                                const int* __restrict__ queries,
                                float* __restrict__ x) {
  int i = blockIdx.x * blockDim.x + threadIdx.x;
  if (i >= BATCH_ * DIM_) return;
  int m = i >> 7, k = i & 127;
  x[i] = emb[(size_t)queries[m] * DIM_ + k];
}

__global__ void scatter_ones_kernel(const int* __restrict__ heads,
                                    float* __restrict__ mem) {
  int b = threadIdx.x;
  mem[(size_t)heads[b] * BATCH_ + b] = 1.0f;
}

// ---------------------------------------------------------------------------
// Input projection: XW[rd] = x @ Wih[rd]^T + bih + bhh   (6 GEMMs, WMMA)
// grid (64, 6), block 128 -> 256 waves per rd, one 16x16 tile per wave
// ---------------------------------------------------------------------------
__global__ void xw_gemm_kernel(const float* __restrict__ x,
                               const float* __restrict__ Wih_f,
                               const float* __restrict__ Wih_b,
                               const float* __restrict__ bih_f,
                               const float* __restrict__ bhh_f,
                               const float* __restrict__ bih_b,
                               const float* __restrict__ bhh_b,
                               float* __restrict__ XW) {
  const int rd = blockIdx.y, r = rd >> 1, dir = rd & 1;
  const float* W  = (dir ? Wih_b : Wih_f) + (size_t)r * G4H * DIM_;
  const float* bi = (dir ? bih_b : bih_f) + (size_t)r * G4H;
  const float* bh = (dir ? bhh_b : bhh_f) + (size_t)r * G4H;

  const int lane = threadIdx.x & 31;
  const int tile = (blockIdx.x * blockDim.x + threadIdx.x) >> 5;  // 0..255
  const int tm = tile >> 5, tn = tile & 31;

  v8f acc = {};
  #pragma unroll
  for (int kb = 0; kb < DIM_ / 32; ++kb) {
    v16h a = load_a_frag_f32(x, DIM_, tm * 16, kb * 32, lane);
    v16h b = load_bT_frag_f32(W, DIM_, tn * 16, kb * 32, lane);
    acc = __builtin_amdgcn_wmma_f32_16x16x32_f16(false, a, false, b,
                                                 (short)0, acc, false, false);
  }
  float* out = XW + (size_t)rd * BATCH_ * G4H;
  const int mbase = tm * 16 + ((lane >> 4) << 3);
  const int n     = tn * 16 + (lane & 15);
  const float bias = bi[n] + bh[n];
  #pragma unroll
  for (int v = 0; v < 8; ++v)
    out[(size_t)(mbase + v) * G4H + n] = acc[v] + bias;
}

// ---------------------------------------------------------------------------
// Sequential LSTM (one workgroup per rank/direction; both directions run as
// forward recurrences since the input is time-constant; backward outputs are
// consumed reversed).  h lives in LDS; z = h @ Whh^T via WMMA.
// grid 6, block 512 (16 waves, 16 tiles each)
// ---------------------------------------------------------------------------
__global__ void lstm_kernel(const float* __restrict__ XW,
                            const float* __restrict__ Whh_f,
                            const float* __restrict__ Whh_b,
                            float* __restrict__ zbuf,
                            float* __restrict__ cbuf,
                            float* __restrict__ HS) {
  __shared__ float h_s[BATCH_ * HID_];  // 64 KB
  const int rd = blockIdx.x, r = rd >> 1, dir = rd & 1;
  const float* W  = (dir ? Whh_b : Whh_f) + (size_t)r * G4H * HID_;
  const float* xw = XW   + (size_t)rd * BATCH_ * G4H;
  float* zb       = zbuf + (size_t)rd * BATCH_ * G4H;
  float* cb       = cbuf + (size_t)rd * BATCH_ * HID_;

  const int tid = threadIdx.x;
  for (int i = tid; i < BATCH_ * HID_; i += blockDim.x) { h_s[i] = 0.0f; cb[i] = 0.0f; }
  __threadfence();
  __syncthreads();

  const int wave = tid >> 5, lane = tid & 31;

  for (int t = 0; t < NSTEPS; ++t) {
    // z = h @ Whh^T + xw
    for (int i = 0; i < 16; ++i) {
      const int tile = wave * 16 + i;
      const int tm = tile >> 5, tn = tile & 31;
      v8f acc = {};
      #pragma unroll
      for (int kb = 0; kb < HID_ / 32; ++kb) {
        v16h a = load_a_frag_f32(h_s, HID_, tm * 16, kb * 32, lane);
        v16h b = load_bT_frag_f32(W, HID_, tn * 16, kb * 32, lane);
        acc = __builtin_amdgcn_wmma_f32_16x16x32_f16(false, a, false, b,
                                                     (short)0, acc, false, false);
      }
      const int mbase = tm * 16 + ((lane >> 4) << 3);
      const int n     = tn * 16 + (lane & 15);
      #pragma unroll
      for (int v = 0; v < 8; ++v)
        zb[(size_t)(mbase + v) * G4H + n] =
            acc[v] + xw[(size_t)(mbase + v) * G4H + n];
    }
    __threadfence();
    __syncthreads();

    // gate update: i,f,g,o order (PyTorch)
    float* hs_out = HS + (size_t)(rd * NSTEPS + t) * BATCH_ * HID_;
    for (int idx = tid; idx < BATCH_ * HID_; idx += blockDim.x) {
      const int b = idx >> 7, j = idx & 127;
      const float zi = zb[(size_t)b * G4H + j];
      const float zf = zb[(size_t)b * G4H + HID_ + j];
      const float zg = zb[(size_t)b * G4H + 2 * HID_ + j];
      const float zo = zb[(size_t)b * G4H + 3 * HID_ + j];
      const float c = sigmoidf_(zf) * cb[idx] + sigmoidf_(zi) * tanhf(zg);
      const float h = sigmoidf_(zo) * tanhf(c);
      cb[idx] = c;
      h_s[idx] = h;
      hs_out[idx] = h;
    }
    __threadfence();
    __syncthreads();
  }
}

// ---------------------------------------------------------------------------
// trans = concat(hf[t], hb[S-1-t]) @ W0 + b0 ; softmax over 13 ops
// grid RANK*NSTEPS, block 128 (one thread per batch element)
// ---------------------------------------------------------------------------
__global__ void attn_kernel(const float* __restrict__ HS,
                            const float* __restrict__ W0,
                            const float* __restrict__ b0,
                            float* __restrict__ attn) {
  const int r = blockIdx.x / NSTEPS, t = blockIdx.x % NSTEPS, b = threadIdx.x;
  const float* hf = HS + ((size_t)((r * 2 + 0) * NSTEPS + t) * BATCH_ + b) * HID_;
  const float* hb = HS + ((size_t)((r * 2 + 1) * NSTEPS + (NSTEPS - 1 - t)) * BATCH_ + b) * HID_;

  float acc[NOP1];
  #pragma unroll
  for (int o = 0; o < NOP1; ++o) acc[o] = b0[o];
  for (int k = 0; k < HID_; ++k) {
    const float f = hf[k];
    #pragma unroll
    for (int o = 0; o < NOP1; ++o) acc[o] += f * W0[k * NOP1 + o];
  }
  for (int k = 0; k < HID_; ++k) {
    const float f = hb[k];
    #pragma unroll
    for (int o = 0; o < NOP1; ++o) acc[o] += f * W0[(HID_ + k) * NOP1 + o];
  }
  float mx = acc[0];
  #pragma unroll
  for (int o = 1; o < NOP1; ++o) mx = fmaxf(mx, acc[o]);
  float s = 0.0f;
  #pragma unroll
  for (int o = 0; o < NOP1; ++o) { acc[o] = __expf(acc[o] - mx); s += acc[o]; }
  const float inv = 1.0f / s;
  float* arow = attn + (size_t)(r * NSTEPS + t) * NOP1 * BATCH_;
  #pragma unroll
  for (int o = 0; o < NOP1; ++o) arow[(size_t)o * BATCH_ + b] = acc[o] * inv;
}

// ---------------------------------------------------------------------------
// memB[e][b] = memA[e][b] * attn_identity[b]   (full overwrite before atomics)
// ---------------------------------------------------------------------------
__global__ void scale_identity_kernel(const float* __restrict__ memA,
                                      float* __restrict__ memB,
                                      const float* __restrict__ a12) {
  const int i = blockIdx.x * blockDim.x + threadIdx.x;  // float4 index
  if (i >= NENT * BATCH_ / 4) return;
  const int p = i * 4;
  const float4 a = *(const float4*)(a12 + (p & 127));
  const float4 m = *(const float4*)(memA + (size_t)p);
  float4 o;
  o.x = m.x * a.x; o.y = m.y * a.y; o.z = m.z * a.z; o.w = m.w * a.w;
  *(float4*)(memB + (size_t)p) = o;
}

// ---------------------------------------------------------------------------
// SpMM scatter: for each edge e of op: memB[dst][b] += attn[op][b]*val*memA[src][b]
// One wave per edge; 32 lanes x float4 cover BATCH=128 (contiguous 512B rows).
// Edge indices fetched wave-uniformly (readfirstlane -> SMEM path), payload is
// a single coalesced 512B gather + 4 native f32 atomics per lane into L2.
// grid (1024, NOPS), block 256
// ---------------------------------------------------------------------------
__global__ void spmm_kernel(const float* __restrict__ memA,
                            float* __restrict__ memB,
                            const int* __restrict__ edge_src,
                            const int* __restrict__ edge_dst,
                            const float* __restrict__ edge_val,
                            const float* __restrict__ attn_t) {
  const int op = blockIdx.y;
  const int*   s = edge_src + (size_t)op * NNZ_;
  const int*   d = edge_dst + (size_t)op * NNZ_;
  const float* v = edge_val + (size_t)op * NNZ_;
  const float* arow = attn_t + (size_t)op * BATCH_;

  const int lane = threadIdx.x & 31;
  // wave index as a provably-uniform (SGPR) value so edge metadata loads
  // become scalar loads
  const int wave_in_blk = __builtin_amdgcn_readfirstlane(threadIdx.x >> 5);
  const int wpb    = blockDim.x >> 5;
  const int wave   = blockIdx.x * wpb + wave_in_blk;
  const int nwaves = gridDim.x * wpb;
  const float4 a4  = *(const float4*)(arow + lane * 4);

  for (int e = wave; e < NNZ_; e += nwaves) {
    const int se = s[e];
    const int de = d[e];
    const float ve = v[e];
    const int en = e + nwaves;
    if (en < NNZ_)  // prefetch next gather row -> global_prefetch_b8
      __builtin_prefetch(memA + (size_t)s[en] * BATCH_ + lane * 4, 0, 1);
    const float4 m = *(const float4*)(memA + (size_t)se * BATCH_ + lane * 4);
    float* o = memB + (size_t)de * BATCH_ + lane * 4;
    unsafeAtomicAdd(o + 0, m.x * a4.x * ve);   // global_atomic_add_f32
    unsafeAtomicAdd(o + 1, m.y * a4.y * ve);
    unsafeAtomicAdd(o + 2, m.z * a4.z * ve);
    unsafeAtomicAdd(o + 3, m.w * a4.w * ve);
  }
}

// ---------------------------------------------------------------------------
// norm[b] += sum_e mem[e][b]  (coalesced column sums)
// ---------------------------------------------------------------------------
__global__ void colsum_kernel(const float* __restrict__ mem,
                              float* __restrict__ norm) {
  const int b = threadIdx.x;
  const int e0 = blockIdx.x * 128;
  const int e1 = min(e0 + 128, NENT);
  float s = 0.0f;
  for (int e = e0; e < e1; ++e) s += mem[(size_t)e * BATCH_ + b];
  unsafeAtomicAdd(norm + b, s);
}

// ---------------------------------------------------------------------------
// out[b][e] += mem[e][b] / max(norm[b],1e-20)  via padded-LDS transpose.
// Tile staging goes through the CDNA5 async global->LDS path (ASYNCcnt) when
// the toolchain exposes it; falls back to plain loads otherwise.
// grid NENT/64, block 256
// ---------------------------------------------------------------------------
__global__ void addout_kernel(const float* __restrict__ mem,
                              const float* __restrict__ norm,
                              float* __restrict__ out) {
  __shared__ float tile[64 * 129];
  const int e0 = blockIdx.x * 64;
#if HAVE_ASYNC_LDS
  for (int idx = threadIdx.x; idx < 64 * BATCH_; idx += blockDim.x) {
    const int i = idx >> 7, b = idx & 127;
    __builtin_amdgcn_global_load_async_to_lds_b32(
        (gptr_i32)(mem + (size_t)(e0 + i) * BATCH_ + b),
        (lptr_i32)&tile[i * 129 + b],
        /*offset=*/0, /*cpol=*/0);
  }
  __builtin_amdgcn_s_wait_asynccnt(0);
  __syncthreads();
#else
  for (int idx = threadIdx.x; idx < 64 * BATCH_; idx += blockDim.x) {
    const int i = idx >> 7, b = idx & 127;
    tile[i * 129 + b] = mem[(size_t)(e0 + i) * BATCH_ + b];
  }
  __syncthreads();
#endif
  for (int idx = threadIdx.x; idx < 64 * BATCH_; idx += blockDim.x) {
    const int b = idx >> 6, i = idx & 63;
    const float inv = 1.0f / fmaxf(norm[b], 1e-20f);
    out[(size_t)b * NENT + e0 + i] += tile[i * 129 + b] * inv;
  }
}

// ---------------------------------------------------------------------------
extern "C" void kernel_launch(void* const* d_in, const int* in_sizes, int n_in,
                              void* d_out, int out_size, void* d_ws, size_t ws_size,
                              hipStream_t stream) {
  (void)in_sizes; (void)n_in; (void)out_size; (void)ws_size;
  const int*   queries = (const int*)d_in[0];
  const int*   heads   = (const int*)d_in[1];
  const int*   e_src   = (const int*)d_in[2];
  const int*   e_dst   = (const int*)d_in[3];
  const float* e_val   = (const float*)d_in[4];
  const float* emb     = (const float*)d_in[5];
  const float* Wih_f   = (const float*)d_in[6];
  const float* Whh_f   = (const float*)d_in[7];
  const float* bih_f   = (const float*)d_in[8];
  const float* bhh_f   = (const float*)d_in[9];
  const float* Wih_b   = (const float*)d_in[10];
  const float* Whh_b   = (const float*)d_in[11];
  const float* bih_b   = (const float*)d_in[12];
  const float* bhh_b   = (const float*)d_in[13];
  const float* W0      = (const float*)d_in[14];
  const float* b0      = (const float*)d_in[15];
  float* out = (float*)d_out;

  char* ws = (char*)d_ws;
  size_t off = 0;
  auto alloc = [&](size_t nfloats) {
    float* p = (float*)(ws + off);
    off += nfloats * sizeof(float);
    return p;
  };
  float* x    = alloc((size_t)BATCH_ * DIM_);
  float* XW   = alloc(6ull * BATCH_ * G4H);
  float* zbuf = alloc(6ull * BATCH_ * G4H);
  float* cbuf = alloc(6ull * BATCH_ * HID_);
  float* HS   = alloc((size_t)RANK_ * 2 * NSTEPS * BATCH_ * HID_);
  float* attn = alloc((size_t)RANK_ * NSTEPS * NOP1 * BATCH_);
  float* norm = alloc(BATCH_);
  off = (off + 255) & ~(size_t)255;
  float* memA = (float*)(ws + off); off += (size_t)NENT * BATCH_ * sizeof(float);
  float* memB = (float*)(ws + off); off += (size_t)NENT * BATCH_ * sizeof(float);

  const int NTOT = NENT * BATCH_;  // 5,120,000 (divisible by 4)

  // ---- phase A: zero output, LSTM + attention -----------------------------
  zero4_kernel<<<(NTOT / 4 + 255) / 256, 256, 0, stream>>>((float4*)out, NTOT / 4);
  gather_x_kernel<<<(BATCH_ * DIM_ + 255) / 256, 256, 0, stream>>>(emb, queries, x);
  xw_gemm_kernel<<<dim3(64, 6), 128, 0, stream>>>(x, Wih_f, Wih_b,
                                                  bih_f, bhh_f, bih_b, bhh_b, XW);
  lstm_kernel<<<6, 512, 0, stream>>>(XW, Whh_f, Whh_b, zbuf, cbuf, HS);
  attn_kernel<<<RANK_ * NSTEPS, BATCH_, 0, stream>>>(HS, W0, b0, attn);

  // ---- phase B: graph propagation (L2-resident (E,B) layout) --------------
  for (int r = 0; r < RANK_; ++r) {
    zero4_kernel<<<(NTOT / 4 + 255) / 256, 256, 0, stream>>>((float4*)memA, NTOT / 4);
    scatter_ones_kernel<<<1, BATCH_, 0, stream>>>(heads, memA);
    float* A = memA;
    float* B = memB;
    for (int t = 0; t < NSTEPS; ++t) {
      const float* attn_t = attn + (size_t)(r * NSTEPS + t) * NOP1 * BATCH_;
      scale_identity_kernel<<<(NTOT / 4 + 255) / 256, 256, 0, stream>>>(
          A, B, attn_t + (size_t)NOPS * BATCH_);
      spmm_kernel<<<dim3(1024, NOPS), 256, 0, stream>>>(A, B, e_src, e_dst,
                                                        e_val, attn_t);
      float* tmp = A; A = B; B = tmp;
    }
    zero4_kernel<<<1, 32, 0, stream>>>((float4*)norm, BATCH_ / 4);
    colsum_kernel<<<(NENT + 127) / 128, 128, 0, stream>>>(A, norm);
    addout_kernel<<<NENT / 64, 256, 0, stream>>>(A, norm, out);
  }
}
